// Text2Mel_59940563583453
// MI455X (gfx1250) — matrix operand order repaired and checked by reading, not compiled, not run on hardware
//
#include <hip/hip_runtime.h>
#include <hip/hip_bf16.h>
#include <math.h>

// ---------------------------------------------------------------------------
// Types for CDNA5 (gfx1250) WMMA: wave32, 16x16x32 f16 -> f32
// ---------------------------------------------------------------------------
typedef __attribute__((ext_vector_type(16))) _Float16 v16h;
typedef __attribute__((ext_vector_type(8)))  _Float16 v8h;
typedef __attribute__((ext_vector_type(8)))  float    v8f;

#define DEV __device__ __forceinline__

DEV v16h cat8(v8h a, v8h b) {
  return __builtin_shufflevector(a, b, 0,1,2,3,4,5,6,7,8,9,10,11,12,13,14,15);
}
DEV v8f wmma_f16(v16h a, v16h b, v8f c) {
  return __builtin_amdgcn_wmma_f32_16x16x32_f16(false, a, false, b, (short)0, c, false, false);
}
DEV float sigf(float x) { return 1.f / (1.f + expf(-x)); }

// ---------------------------------------------------------------------------
// Problem constants
// ---------------------------------------------------------------------------
#define BB 32
#define NN 256
#define TT 1024
#define FMEL 80
#define PAD 64                 // zero guard rows each side (max reach = 54)
#define TPT (NN + 2 * PAD)     // padded text length  = 384
#define TPA (TT + 2 * PAD)     // padded audio length = 1152

// ---------------------------------------------------------------------------
// Dilated conv1d as WMMA GEMM, 2x2 register tiling (32 time x 32 oc / wave).
// x: [B][TP][CIN] f16, time rows padded with PAD zero rows each side.
// w: [ocPad][K][CIN] f16 (ic and oc zero-padded).
// All fragment loads unconditional (pads absorb the conv reach).
// outMode 0: f16 -> y interior rows of [B][TP][CoutStride]
// outMode 1: f32 logit+sigmoid -> [B][CoutTotal][Tlen] (reference layout)
// ---------------------------------------------------------------------------
template <int CIN>
__global__ __launch_bounds__(32)
void k_conv1d_wmma(const _Float16* __restrict__ x, const _Float16* __restrict__ w,
                   const float* __restrict__ bias, _Float16* __restrict__ y,
                   float* __restrict__ ylogit, float* __restrict__ ysig,
                   int Tlen, int TP, int CoutStride, int CoutTotal,
                   int K, int dil, int causal, int act, int outMode)
{
  const int t0   = blockIdx.x << 5;
  const int oc0  = blockIdx.y << 5;
  const int b    = blockIdx.z;
  const int lane = threadIdx.x & 31;
  const int half = lane >> 4;
  const int l15  = lane & 15;

  v8f acc00 = {}, acc01 = {}, acc10 = {}, acc11 = {};
  const int padL = causal ? (K - 1) * dil : (((K - 1) * dil) >> 1);

  for (int j = 0; j < K; ++j) {
    const int shift = j * dil - padL;                      // in [-54, +27]
    const _Float16* ar0 = x + ((size_t)b * TP + PAD + t0 + l15 + shift) * CIN + half * 8;
    const _Float16* ar1 = ar0 + (size_t)16 * CIN;
    const _Float16* wr0 = w + ((size_t)(oc0 + l15) * K + j) * CIN + half * 8;
    const _Float16* wr1 = wr0 + (size_t)16 * K * CIN;
    if (j + 1 < K) __builtin_prefetch(ar0 + (size_t)dil * CIN, 0, 1);
#pragma unroll
    for (int ic = 0; ic < CIN; ic += 32) {
      v16h a0 = cat8(*(const v8h*)(ar0 + ic), *(const v8h*)(ar0 + ic + 16));
      v16h a1 = cat8(*(const v8h*)(ar1 + ic), *(const v8h*)(ar1 + ic + 16));
      v16h b0 = cat8(*(const v8h*)(wr0 + ic), *(const v8h*)(wr0 + ic + 16));
      v16h b1 = cat8(*(const v8h*)(wr1 + ic), *(const v8h*)(wr1 + ic + 16));
      acc00 = wmma_f16(a0, b0, acc00);
      acc01 = wmma_f16(a0, b1, acc01);
      acc10 = wmma_f16(a1, b0, acc10);
      acc11 = wmma_f16(a1, b1, acc11);
    }
  }

  auto store_tile = [&](const v8f& acc, int rt, int occ) {
    if (occ >= CoutTotal) return;
    const float bb = bias[occ];
    if (outMode == 0) {
      _Float16* yb = y + ((size_t)b * TP + PAD + t0 + rt * 16 + half * 8) * CoutStride + occ;
#pragma unroll
      for (int r = 0; r < 8; ++r) {
        float v = acc[r] + bb;
        if (act) v = fmaxf(v, 0.f);
        yb[(size_t)r * CoutStride] = (_Float16)v;
      }
    } else {
      const size_t base = ((size_t)b * CoutTotal + occ) * (size_t)Tlen
                        + t0 + rt * 16 + half * 8;
#pragma unroll
      for (int r = 0; r < 8; ++r) {
        const float v = acc[r] + bb;
        ylogit[base + r] = v;
        ysig[base + r]   = sigf(v);
      }
    }
  };
  store_tile(acc00, 0, oc0 + l15);
  store_tile(acc01, 0, oc0 + 16 + l15);
  store_tile(acc10, 1, oc0 + l15);
  store_tile(acc11, 1, oc0 + 16 + l15);
}

// ---------------------------------------------------------------------------
// Attention scores: A[b][n][t] = (1/16) * sum_d K[b][n][d] * Q[b][t][d]
// X (text out, padded [B][TPT][512], K = ch 0..255); Q (padded [B][TPA][256])
// 2x2 tiling: 32 n x 32 t per wave.
// ---------------------------------------------------------------------------
__global__ __launch_bounds__(32)
void k_attn_scores(const _Float16* __restrict__ X, const _Float16* __restrict__ Q,
                   float* __restrict__ Araw)
{
  const int t0 = blockIdx.x << 5, n0 = blockIdx.y << 5, b = blockIdx.z;
  const int lane = threadIdx.x & 31, half = lane >> 4, l15 = lane & 15;
  v8f a00 = {}, a01 = {}, a10 = {}, a11 = {};
  const _Float16* ar0 = X + ((size_t)b * TPT + PAD + n0 + l15) * 512 + half * 8;
  const _Float16* ar1 = ar0 + (size_t)16 * 512;
  const _Float16* br0 = Q + ((size_t)b * TPA + PAD + t0 + l15) * 256 + half * 8;
  const _Float16* br1 = br0 + (size_t)16 * 256;
#pragma unroll
  for (int d = 0; d < 256; d += 32) {
    v16h an0 = cat8(*(const v8h*)(ar0 + d), *(const v8h*)(ar0 + d + 16));
    v16h an1 = cat8(*(const v8h*)(ar1 + d), *(const v8h*)(ar1 + d + 16));
    v16h bt0 = cat8(*(const v8h*)(br0 + d), *(const v8h*)(br0 + d + 16));
    v16h bt1 = cat8(*(const v8h*)(br1 + d), *(const v8h*)(br1 + d + 16));
    a00 = wmma_f16(an0, bt0, a00);
    a01 = wmma_f16(an0, bt1, a01);
    a10 = wmma_f16(an1, bt0, a10);
    a11 = wmma_f16(an1, bt1, a11);
  }
  auto st = [&](const v8f& acc, int nt, int tt) {
#pragma unroll
    for (int r = 0; r < 8; ++r) {
      const int n = n0 + nt * 16 + r + half * 8;
      const int t = t0 + tt * 16 + l15;
      Araw[((size_t)b * NN + n) * TT + t] = acc[r] * 0.0625f;
    }
  };
  st(a00, 0, 0); st(a01, 0, 1); st(a10, 1, 0); st(a11, 1, 1);
}

// ---------------------------------------------------------------------------
// R[b][t][d] = sum_n At[b][t][n] * V[b][n][d]  -> RP interior rows, ch 0..255
// At: [B][T][256] f16 (unpadded); Vt: [B][256][256] f16 (d rows, n contig)
// ---------------------------------------------------------------------------
__global__ __launch_bounds__(32)
void k_attn_apply(const _Float16* __restrict__ At, const _Float16* __restrict__ Vt,
                  _Float16* __restrict__ RP)
{
  const int t0 = blockIdx.x << 5, d0 = blockIdx.y << 5, b = blockIdx.z;
  const int lane = threadIdx.x & 31, half = lane >> 4, l15 = lane & 15;
  v8f a00 = {}, a01 = {}, a10 = {}, a11 = {};
  const _Float16* ar0 = At + ((size_t)b * TT + t0 + l15) * NN + half * 8;
  const _Float16* ar1 = ar0 + (size_t)16 * NN;
  const _Float16* br0 = Vt + ((size_t)b * 256 + d0 + l15) * NN + half * 8;
  const _Float16* br1 = br0 + (size_t)16 * NN;
#pragma unroll
  for (int n = 0; n < NN; n += 32) {
    v16h at0 = cat8(*(const v8h*)(ar0 + n), *(const v8h*)(ar0 + n + 16));
    v16h at1 = cat8(*(const v8h*)(ar1 + n), *(const v8h*)(ar1 + n + 16));
    v16h bd0 = cat8(*(const v8h*)(br0 + n), *(const v8h*)(br0 + n + 16));
    v16h bd1 = cat8(*(const v8h*)(br1 + n), *(const v8h*)(br1 + n + 16));
    a00 = wmma_f16(at0, bd0, a00);
    a01 = wmma_f16(at0, bd1, a01);
    a10 = wmma_f16(at1, bd0, a10);
    a11 = wmma_f16(at1, bd1, a11);
  }
  auto st = [&](const v8f& acc, int tt, int dt) {
#pragma unroll
    for (int r = 0; r < 8; ++r) {
      const int t = t0 + tt * 16 + r + half * 8;
      RP[((size_t)b * TPA + PAD + t) * 512 + d0 + dt * 16 + l15] = (_Float16)acc[r];
    }
  };
  st(a00, 0, 0); st(a01, 0, 1); st(a10, 1, 0); st(a11, 1, 1);
}

// ---------------------------------------------------------------------------
// monotonic_fix: sequential scan along T per batch (matches reference scan).
// ---------------------------------------------------------------------------
__global__ __launch_bounds__(256)
void k_monotonic(float* __restrict__ A, const int* __restrict__ flag)
{
  if (*flag == 0) return;
  const int b = blockIdx.x;
  const int n = threadIdx.x;
  __shared__ float sv[256];
  __shared__ int   si[256];
  __shared__ int   sprev, scond, sforced;
  if (n == 0) sprev = 0;
  float* Ab = A + (size_t)b * NN * TT;
  for (int t = 0; t < TT; ++t) {
    sv[n] = Ab[(size_t)n * TT + t];
    si[n] = n;
    __syncthreads();
    for (int s = 128; s > 0; s >>= 1) {
      if (n < s) {
        const float v2 = sv[n + s]; const int i2 = si[n + s];
        if (v2 > sv[n] || (v2 == sv[n] && i2 < si[n])) { sv[n] = v2; si[n] = i2; }
      }
      __syncthreads();
    }
    if (n == 0) {
      const int cur = si[0];
      const int prev = sprev;
      int d = cur - prev; if (d < 0) d = -d;
      const int cond = (d > 3);
      int forced = prev + 1; if (forced > NN - 1) forced = NN - 1;
      scond = cond; sforced = forced;
      sprev = cond ? forced : cur;
    }
    __syncthreads();
    if (scond) Ab[(size_t)n * TT + t] = (n == sforced) ? 1.0f : -1e9f;
    __syncthreads();
  }
}

// ---------------------------------------------------------------------------
// Softmax over n. One wave per (b,t) column; emits f32 A and f16 At[B][T][N].
// ---------------------------------------------------------------------------
__global__ __launch_bounds__(256)
void k_softmax(const float* __restrict__ Araw, float* __restrict__ Aout,
               _Float16* __restrict__ At16)
{
  const int wid = threadIdx.x >> 5, lane = threadIdx.x & 31;
  const int t = blockIdx.x * 8 + wid;
  const int b = blockIdx.y;
  const float* col = Araw + (size_t)b * NN * TT + t;
  float v[8];
  float m = -3.4e38f;
#pragma unroll
  for (int i = 0; i < 8; ++i) { v[i] = col[(size_t)(lane + 32 * i) * TT]; m = fmaxf(m, v[i]); }
  for (int o = 16; o > 0; o >>= 1) m = fmaxf(m, __shfl_xor(m, o, 32));
  float s = 0.f;
#pragma unroll
  for (int i = 0; i < 8; ++i) { v[i] = expf(v[i] - m); s += v[i]; }
  for (int o = 16; o > 0; o >>= 1) s += __shfl_xor(s, o, 32);
  const float inv = 1.f / s;
#pragma unroll
  for (int i = 0; i < 8; ++i) {
    const int n = lane + 32 * i;
    const float p = v[i] * inv;
    Aout[(size_t)b * NN * TT + (size_t)n * TT + t] = p;
    At16[((size_t)b * TT + t) * NN + n] = (_Float16)p;
  }
}

// ---------------------------------------------------------------------------
// Elementwise helpers (all padded-layout aware)
// ---------------------------------------------------------------------------
__global__ void k_zeropad(_Float16* __restrict__ buf, int TP, int C)
{
  int i = blockIdx.x * blockDim.x + threadIdx.x;
  const int tot = BB * 2 * PAD * C;
  if (i >= tot) return;
  const int c = i % C, r = (i / C) % (2 * PAD), b = i / (2 * PAD * C);
  const int row = (r < PAD) ? r : (TP - 2 * PAD + r);
  buf[((size_t)b * TP + row) * C + c] = (_Float16)0.f;
}

__global__ void k_highway_gate(const _Float16* __restrict__ h, const _Float16* __restrict__ xin,
                               _Float16* __restrict__ out, int total, int C, int Len, int TP)
{
  int i = blockIdx.x * blockDim.x + threadIdx.x;
  if (i >= total) return;
  const int c = i % C, t = (i / C) % Len, b = i / (C * Len);
  const size_t row = (size_t)b * TP + PAD + t;
  const float h1 = (float)h[row * 2 * C + c];
  const float h2 = (float)h[row * 2 * C + C + c];
  const float xv = (float)xin[row * C + c];
  const float g = sigf(h1);
  out[row * C + c] = (_Float16)(g * h2 + (1.f - g) * xv);
}

__global__ void k_embed(const int* __restrict__ L, const float* __restrict__ emb,
                        _Float16* __restrict__ out)
{
  const int bn = blockIdx.x, e = threadIdx.x;
  const int b = bn >> 8, n = bn & 255;
  out[((size_t)b * TPT + PAD + n) * 128 + e] = (_Float16)emb[(size_t)L[bn] * 128 + e];
}

__global__ void k_spack(const float* __restrict__ S, _Float16* __restrict__ S16, int total)
{
  int i = blockIdx.x * blockDim.x + threadIdx.x;
  if (i >= total) return;
  const int c = i % 96, t = (i / 96) % TT, b = i / (96 * TT);
  const float v = (c < FMEL) ? S[((size_t)b * FMEL + c) * TT + t] : 0.f;
  S16[((size_t)b * TPA + PAD + t) * 96 + c] = (_Float16)v;
}

// w (OIW f32, [oc][ic][k]) -> wout [ocPad][k][icPad] f16 (zero-padded); bias too.
__global__ void k_packw(const float* __restrict__ w, const float* __restrict__ bsrc,
                        _Float16* __restrict__ wout, float* __restrict__ bout,
                        int oc, int ocPad, int ic, int icPad, int k)
{
  int i = blockIdx.x * blockDim.x + threadIdx.x;
  if (i < ocPad) bout[i] = (i < oc) ? bsrc[i] : 0.f;
  const int tot = ocPad * k * icPad;
  if (i >= tot) return;
  const int c = i % icPad, j = (i / icPad) % k, o = i / (icPad * k);
  wout[i] = (_Float16)((o < oc && c < ic) ? w[((size_t)o * ic + c) * k + j] : 0.f);
}

// Vt[b][d][n] = X[b][n][256+d]   (X padded text buffer, stride 512)
__global__ void k_vpack(const _Float16* __restrict__ X, _Float16* __restrict__ Vt, int total)
{
  int i = blockIdx.x * blockDim.x + threadIdx.x;
  if (i >= total) return;
  const int n = i % 256, d = (i / 256) % 256, b = i / 65536;
  Vt[i] = X[((size_t)b * TPT + PAD + n) * 512 + 256 + d];
}

// RP[b][PAD+t][256+c] = Q[b][PAD+t][c]
__global__ void k_concatq(const _Float16* __restrict__ Qf, _Float16* __restrict__ RP, int total)
{
  int i = blockIdx.x * blockDim.x + threadIdx.x;
  if (i >= total) return;
  const int c = i % 256, t = (i / 256) % TT, b = i / (256 * TT);
  const size_t row = (size_t)b * TPA + PAD + t;
  RP[row * 512 + 256 + c] = Qf[row * 256 + c];
}

// ---------------------------------------------------------------------------
// Host orchestration
// ---------------------------------------------------------------------------
extern "C" void kernel_launch(void* const* d_in, const int* in_sizes, int n_in,
                              void* d_out, int out_size, void* d_ws, size_t ws_size,
                              hipStream_t stream)
{
  (void)in_sizes; (void)n_in; (void)out_size; (void)ws_size;

  const int*   L    = (const int*)d_in[0];
  const float* S    = (const float*)d_in[1];
  const int*   mono = (const int*)d_in[2];
  const float* emb  = (const float*)d_in[3];

  struct LD { int widx, ic, icPad, oc, ocPad, k, dil, causal, act; };
  static const int thk[12] = {3,3,3,3,3,3,3,3,3,3,1,1};
  static const int thd[12] = {1,3,9,27,1,3,9,27,1,1,1,1};
  static const int ahd[10] = {1,3,9,27,1,3,9,27,3,3};
  static const int dhk[7]  = {3,3,3,3,3,3,1};
  static const int dhd[7]  = {1,3,9,27,1,1,1};
  LD ld[37]; int nl = 0;
  ld[nl++] = {4, 128, 128, 512, 512, 1, 1, 0, 1};
  ld[nl++] = {6, 512, 512, 512, 512, 1, 1, 0, 0};
  for (int i = 0; i < 12; ++i) ld[nl++] = {8 + 2*i, 512, 512, 1024, 1024, thk[i], thd[i], 0, 0};
  ld[nl++] = {32, 80,  96,  256, 256, 1, 1, 1, 1};
  ld[nl++] = {34, 256, 256, 256, 256, 1, 1, 1, 1};
  ld[nl++] = {36, 256, 256, 256, 256, 1, 1, 1, 0};
  for (int i = 0; i < 10; ++i) ld[nl++] = {38 + 2*i, 256, 256, 512, 512, 3, ahd[i], 1, 0};
  ld[nl++] = {58, 512, 512, 256, 256, 1, 1, 1, 0};
  for (int i = 0; i < 7; ++i)  ld[nl++] = {60 + 2*i, 256, 256, 512, 512, dhk[i], dhd[i], 1, 0};
  ld[nl++] = {74, 256, 256, 256, 256, 1, 1, 1, 1};
  ld[nl++] = {76, 256, 256, 80,  96,  1, 1, 1, 0};

  char* ws = (char*)d_ws;
  size_t off = 0;
  auto take = [&](size_t bytes) -> void* {
    void* p = ws + off;
    off = (off + bytes + 255) & ~(size_t)255;
    return p;
  };

  // Weights -> padded f16
  _Float16* Wp[37]; float* Bp[37];
  for (int l = 0; l < 37; ++l) {
    const LD& d = ld[l];
    const int tot = d.ocPad * d.k * d.icPad;
    Wp[l] = (_Float16*)take((size_t)tot * 2);
    Bp[l] = (float*)take((size_t)d.ocPad * 4);
    k_packw<<<(tot + 255) / 256, 256, 0, stream>>>((const float*)d_in[d.widx],
                                                   (const float*)d_in[d.widx + 1],
                                                   Wp[l], Bp[l], d.oc, d.ocPad, d.ic, d.icPad, d.k);
  }

  // Activations (time-padded layouts)
  _Float16* EMB16 = (_Float16*)take((size_t)BB * TPT * 128 * 2);
  _Float16* T0b   = (_Float16*)take((size_t)BB * TPT * 512 * 2);
  _Float16* T1b   = (_Float16*)take((size_t)BB * TPT * 512 * 2);
  _Float16* THb   = (_Float16*)take((size_t)BB * TPT * 1024 * 2);
  _Float16* S16   = (_Float16*)take((size_t)BB * TPA * 96 * 2);
  _Float16* Q0b   = (_Float16*)take((size_t)BB * TPA * 256 * 2);
  _Float16* Q1b   = (_Float16*)take((size_t)BB * TPA * 256 * 2);
  _Float16* QHb   = (_Float16*)take((size_t)BB * TPA * 512 * 2);
  _Float16* Vt    = (_Float16*)take((size_t)BB * 256 * 256 * 2);
  float*    ARAW  = (float*)take((size_t)BB * NN * TT * 4);
  _Float16* At16  = (_Float16*)take((size_t)BB * TT * 256 * 2);
  _Float16* RPb   = (_Float16*)take((size_t)BB * TPA * 512 * 2);
  _Float16* D0b   = (_Float16*)take((size_t)BB * TPA * 256 * 2);
  _Float16* D1b   = (_Float16*)take((size_t)BB * TPA * 256 * 2);
  _Float16* DHb   = (_Float16*)take((size_t)BB * TPA * 512 * 2);

  // Zero the guard bands of buffers read by k>1 convs (every call: deterministic).
  auto zpad = [&](_Float16* buf, int TP, int C) {
    const int tot = BB * 2 * PAD * C;
    k_zeropad<<<(tot + 255) / 256, 256, 0, stream>>>(buf, TP, C);
  };
  zpad(T0b, TPT, 512); zpad(T1b, TPT, 512);
  zpad(Q0b, TPA, 256); zpad(Q1b, TPA, 256);
  zpad(D0b, TPA, 256); zpad(D1b, TPA, 256);

  auto conv = [&](const _Float16* x, _Float16* y, int l, int Tlen, int TP, int outStride,
                  float* ylog, float* ysig) {
    const LD& d = ld[l];
    dim3 g(Tlen / 32, d.ocPad / 32, BB);
    const int om = ylog ? 1 : 0;
    switch (d.icPad) {
      case 96:
        k_conv1d_wmma<96><<<g, 32, 0, stream>>>(x, Wp[l], Bp[l], y, ylog, ysig, Tlen, TP,
                                                outStride, d.oc, d.k, d.dil, d.causal, d.act, om);
        break;
      case 128:
        k_conv1d_wmma<128><<<g, 32, 0, stream>>>(x, Wp[l], Bp[l], y, ylog, ysig, Tlen, TP,
                                                 outStride, d.oc, d.k, d.dil, d.causal, d.act, om);
        break;
      case 256:
        k_conv1d_wmma<256><<<g, 32, 0, stream>>>(x, Wp[l], Bp[l], y, ylog, ysig, Tlen, TP,
                                                 outStride, d.oc, d.k, d.dil, d.causal, d.act, om);
        break;
      default:
        k_conv1d_wmma<512><<<g, 32, 0, stream>>>(x, Wp[l], Bp[l], y, ylog, ysig, Tlen, TP,
                                                 outStride, d.oc, d.k, d.dil, d.causal, d.act, om);
        break;
    }
  };
  auto hwgate = [&](const _Float16* h, const _Float16* xin, _Float16* out, int Len, int TP, int C) {
    const int tot = BB * Len * C;
    k_highway_gate<<<(tot + 255) / 256, 256, 0, stream>>>(h, xin, out, tot, C, Len, TP);
  };

  // ---- Text encoder (N=256, non-causal) ----
  k_embed<<<BB * NN, 128, 0, stream>>>(L, emb, EMB16);
  conv(EMB16, T0b, 0, NN, TPT, 512, nullptr, nullptr);
  conv(T0b, T1b, 1, NN, TPT, 512, nullptr, nullptr);
  _Float16 *cur = T1b, *alt = T0b;
  for (int i = 0; i < 12; ++i) {
    conv(cur, THb, 2 + i, NN, TPT, 1024, nullptr, nullptr);
    hwgate(THb, cur, alt, NN, TPT, 512);
    _Float16* t = cur; cur = alt; alt = t;
  }
  _Float16* Xtext = cur;                                  // [B][TPT][512]: K | V
  { const int tot = BB * 256 * 256;
    k_vpack<<<(tot + 255) / 256, 256, 0, stream>>>(Xtext, Vt, tot); }

  // ---- Audio encoder (T=1024, causal) ----
  { const int tot = BB * TT * 96;
    k_spack<<<(tot + 255) / 256, 256, 0, stream>>>(S, S16, tot); }
  conv(S16, Q0b, 14, TT, TPA, 256, nullptr, nullptr);
  conv(Q0b, Q1b, 15, TT, TPA, 256, nullptr, nullptr);
  conv(Q1b, Q0b, 16, TT, TPA, 256, nullptr, nullptr);
  cur = Q0b; alt = Q1b;
  for (int i = 0; i < 10; ++i) {
    conv(cur, QHb, 17 + i, TT, TPA, 512, nullptr, nullptr);
    hwgate(QHb, cur, alt, TT, TPA, 256);
    _Float16* t = cur; cur = alt; alt = t;
  }
  _Float16* Qf = cur;                                     // [B][TPA][256]

  // ---- Attention ----
  k_attn_scores<<<dim3(TT / 32, NN / 32, BB), 32, 0, stream>>>(Xtext, Qf, ARAW);
  k_monotonic<<<BB, 256, 0, stream>>>(ARAW, mono);
  float* Aout = (float*)d_out + 2 * (size_t)BB * FMEL * TT;
  k_softmax<<<dim3(TT / 8, BB), 256, 0, stream>>>(ARAW, Aout, At16);
  k_attn_apply<<<dim3(TT / 32, 256 / 32, BB), 32, 0, stream>>>(At16, Vt, RPb);
  { const int tot = BB * TT * 256;
    k_concatq<<<(tot + 255) / 256, 256, 0, stream>>>(Qf, RPb, tot); }

  // ---- Audio decoder (T=1024, causal) ----
  conv(RPb, D0b, 27, TT, TPA, 256, nullptr, nullptr);
  cur = D0b; alt = D1b;
  for (int i = 0; i < 7; ++i) {
    conv(cur, DHb, 28 + i, TT, TPA, 512, nullptr, nullptr);
    hwgate(DHb, cur, alt, TT, TPA, 256);
    _Float16* t = cur; cur = alt; alt = t;
  }
  conv(cur, alt, 35, TT, TPA, 256, nullptr, nullptr);     // relu
  float* ylog = (float*)d_out;
  float* ysig = ylog + (size_t)BB * FMEL * TT;
  conv(alt, nullptr, 36, TT, TPA, 0, ylog, ysig);         // Y_logit + Y
}
